// PTv3Attention_38319698215453
// MI455X (gfx1250) — compile-verified, matrix operand order checked by reading, not run on hardware
//
#include <hip/hip_runtime.h>
#include <hip/hip_bf16.h>

// ---------------- types ----------------
typedef __bf16 bf16_t;
typedef __attribute__((ext_vector_type(16))) __bf16 v16bf;
typedef __attribute__((ext_vector_type(8)))  __bf16 v8bf;
typedef __attribute__((ext_vector_type(8)))  float  v8f;

#define BB 8
#define NN 4096
#define CC 256
#define KK 16
#define RPE_DIM 64
#define ROWS (BB * NN)          // 32768
#define SCALE 0.0625f           // 256^-0.5

// =====================================================================
// 0) prep: transpose+convert 4 weight matrices to bf16, collapse W2/b2
// =====================================================================
__global__ __launch_bounds__(256)
void prep_kernel(const float* __restrict__ Wq, const float* __restrict__ Wk,
                 const float* __restrict__ Wv, const float* __restrict__ Wp,
                 const float* __restrict__ W2, const float* __restrict__ b2,
                 bf16_t* __restrict__ Wqt, bf16_t* __restrict__ Wkt,
                 bf16_t* __restrict__ Wvt, bf16_t* __restrict__ Wpt,
                 float* __restrict__ w2s, float* __restrict__ b2s) {
  int t = threadIdx.x;
  const float* src[4] = {Wq, Wk, Wv, Wp};
  bf16_t*      dst[4] = {Wqt, Wkt, Wvt, Wpt};
  for (int w = 0; w < 4; ++w) {
    const float* s = src[w];
    bf16_t* d = dst[w];
    for (int i = t; i < CC * CC; i += 256) {
      int r = i >> 8, c = i & 255;
      d[c * CC + r] = (bf16_t)s[i];          // store W^T in bf16
    }
  }
  if (t < RPE_DIM) {                          // w2s[r] = sum_c W2[r,c]
    float s = 0.f;
    for (int c = 0; c < CC; ++c) s += W2[t * CC + c];
    w2s[t] = s;
  }
  if (t == 0) {                               // b2s = sum_c b2[c]
    float s = 0.f;
    for (int c = 0; c < CC; ++c) s += b2[c];
    b2s[0] = s;
  }
}

// =====================================================================
// 1) fused dual LayerNorm -> bf16 activations for q-path and kv-path
// =====================================================================
__global__ __launch_bounds__(256)
void ln_kernel(const float* __restrict__ X,
               const float* __restrict__ gq, const float* __restrict__ bq,
               const float* __restrict__ gk, const float* __restrict__ bk,
               bf16_t* __restrict__ xq, bf16_t* __restrict__ xkv) {
  __shared__ float s1[256], s2[256];
  int row = blockIdx.x, t = threadIdx.x;
  float x = X[(size_t)row * CC + t];
  s1[t] = x; s2[t] = x * x;
  __syncthreads();
  for (int off = 128; off > 0; off >>= 1) {
    if (t < off) { s1[t] += s1[t + off]; s2[t] += s2[t + off]; }
    __syncthreads();
  }
  float mean = s1[0] * (1.f / CC);
  float var  = s2[0] * (1.f / CC) - mean * mean;
  float rinv = rsqrtf(var + 1e-5f);
  float xn = (x - mean) * rinv;
  xq [(size_t)row * CC + t] = (bf16_t)(xn * gq[t] + bq[t]);
  xkv[(size_t)row * CC + t] = (bf16_t)(xn * gk[t] + bk[t]);
}

// =====================================================================
// 2) bf16 WMMA GEMM:  D[M x 256] = A[M x 256] * Bt^T (+bias)
//    block = 8 waves; wave => 16 rows x 64 cols (4 accumulators)
// =====================================================================
__device__ inline v16bf load_tile16x32(const bf16_t* __restrict__ base,
                                       int ld, int lane) {
  // ISA wave32 16-bit A-fragment layout (16x32, MxK):
  // lanes 0-15: M=lane, K {0..7, 16..23}; lanes 16-31: M=lane-16, K {8..15, 24..31}
  int m = lane & 15;
  int half = lane >> 4;
  const bf16_t* rp = base + (size_t)m * ld;
  v8bf lo = *(const v8bf*)(rp + half * 8);
  v8bf hi = *(const v8bf*)(rp + 16 + half * 8);
  v16bf a;
#pragma unroll
  for (int i = 0; i < 8; ++i) { a[i] = lo[i]; a[i + 8] = hi[i]; }
  return a;
}

template <typename OutT>
__global__ __launch_bounds__(256)
void gemm_bf16_wmma(const bf16_t* __restrict__ A, const bf16_t* __restrict__ Bt,
                    const float* __restrict__ bias, OutT* __restrict__ D) {
  int wave = threadIdx.x >> 5;
  int lane = threadIdx.x & 31;
  int rowBase = blockIdx.x * 128 + wave * 16;
  int nBase   = blockIdx.y * 64;

  v8f acc[4] = {};
  for (int kb = 0; kb < CC; kb += 32) {
    v16bf a = load_tile16x32(A + (size_t)rowBase * CC + kb, CC, lane);
#pragma unroll
    for (int t = 0; t < 4; ++t) {
      v16bf b = load_tile16x32(Bt + (size_t)(nBase + t * 16) * CC + kb, CC, lane);
      acc[t] = __builtin_amdgcn_wmma_f32_16x16x32_bf16(
          false, a, false, b, (short)0, acc[t], false, false);
    }
  }
  // C/D layout: VGPR r -> (M = r | 8+r, N = lane&15)
  int nl = lane & 15;
  int moff = (lane >> 4) * 8;
#pragma unroll
  for (int t = 0; t < 4; ++t) {
    int ncol = nBase + t * 16 + nl;
    float bv = bias ? bias[ncol] : 0.f;
#pragma unroll
    for (int r = 0; r < 8; ++r) {
      D[(size_t)(rowBase + moff + r) * CC + ncol] = (OutT)(acc[t][r] + bv);
    }
  }
}

// =====================================================================
// 3) kNN: per query, top-16 smallest squared distances (LDS-tiled scan)
// =====================================================================
__global__ __launch_bounds__(256)
void knn_kernel(const float* __restrict__ pos, int* __restrict__ idxout) {
  __shared__ float sp[256 * 3];
  int b = blockIdx.y;
  int n = blockIdx.x * 256 + threadIdx.x;
  const float* pb = pos + (size_t)b * NN * 3;
  float px = pb[n * 3], py = pb[n * 3 + 1], pz = pb[n * 3 + 2];
  float dk[16]; int ik[16];
#pragma unroll
  for (int j = 0; j < 16; ++j) { dk[j] = 3.4e38f; ik[j] = 0; }

  for (int tile = 0; tile < NN / 256; ++tile) {
    int cb = tile * 256;
    sp[threadIdx.x]       = pb[cb * 3 + threadIdx.x];
    sp[threadIdx.x + 256] = pb[cb * 3 + 256 + threadIdx.x];
    sp[threadIdx.x + 512] = pb[cb * 3 + 512 + threadIdx.x];
    __syncthreads();
    for (int j = 0; j < 256; ++j) {
      float dx = px - sp[j * 3];
      float dy = py - sp[j * 3 + 1];
      float dz = pz - sp[j * 3 + 2];
      float d = dx * dx + dy * dy + dz * dz;   // diagonal naturally 0
      if (d < dk[15]) {
        dk[15] = d; ik[15] = cb + j;
#pragma unroll
        for (int t = 15; t > 0; --t) {
          if (dk[t] < dk[t - 1]) {
            float td = dk[t]; dk[t] = dk[t - 1]; dk[t - 1] = td;
            int   ti = ik[t]; ik[t] = ik[t - 1]; ik[t - 1] = ti;
          }
        }
      }
    }
    __syncthreads();
  }
  size_t g = (size_t)(b * NN + n) * KK;
#pragma unroll
  for (int j = 0; j < 16; ++j) idxout[g + j] = ik[j];
}

// =====================================================================
// 4) attention: one wave32 per query. RPE collapsed via w2s/b2s.
// =====================================================================
__global__ __launch_bounds__(256)
void attn_kernel(const float* __restrict__ qbuf, const bf16_t* __restrict__ kbuf,
                 const bf16_t* __restrict__ vbuf, const float* __restrict__ pos,
                 const int* __restrict__ idxb, const float* __restrict__ W1,
                 const float* __restrict__ b1, const float* __restrict__ w2s,
                 const float* __restrict__ b2sp, bf16_t* __restrict__ outp) {
  int wave = threadIdx.x >> 5, lane = threadIdx.x & 31;
  int gq = blockIdx.x * 8 + wave;          // 0..32767
  int b = gq >> 12, n = gq & (NN - 1);

  int nbi[16];
  const int* nb = idxb + (size_t)gq * KK;
#pragma unroll
  for (int j = 0; j < 16; ++j) nbi[j] = nb[j];

  float ql[8];
#pragma unroll
  for (int i = 0; i < 8; ++i) ql[i] = qbuf[(size_t)gq * CC + lane + i * 32];

  const float* pb = pos + (size_t)b * NN * 3;
  float px = pb[n * 3], py = pb[n * 3 + 1], pz = pb[n * 3 + 2];
  float b2s = b2sp[0];

  // this lane owns RPE rows r0 = lane, r1 = lane+32
  int r0 = lane, r1 = lane + 32;
  float w0x = W1[r0], w0y = W1[RPE_DIM + r0], w0z = W1[2 * RPE_DIM + r0];
  float w1x = W1[r1], w1y = W1[RPE_DIM + r1], w1z = W1[2 * RPE_DIM + r1];
  float bb0 = b1[r0], bb1 = b1[r1], s0 = w2s[r0], s1 = w2s[r1];

  float logits[16];
#pragma unroll
  for (int j = 0; j < 16; ++j) {
    int m = nbi[j];
    const bf16_t* kr = kbuf + ((size_t)b * NN + m) * CC;
    float part = 0.f;
#pragma unroll
    for (int i = 0; i < 8; ++i) part += ql[i] * (float)kr[lane + i * 32];
    float rx = px - pb[m * 3], ry = py - pb[m * 3 + 1], rz = pz - pb[m * 3 + 2];
    float h0 = fmaxf(rx * w0x + ry * w0y + rz * w0z + bb0, 0.f);
    float h1 = fmaxf(rx * w1x + ry * w1y + rz * w1z + bb1, 0.f);
    part += h0 * s0 + h1 * s1;
#pragma unroll
    for (int off = 16; off > 0; off >>= 1) part += __shfl_xor(part, off, 32);
    logits[j] = (part + b2s) * SCALE;
  }

  float mx = logits[0];
#pragma unroll
  for (int j = 1; j < 16; ++j) mx = fmaxf(mx, logits[j]);
  float wj[16], wsum = 0.f;
#pragma unroll
  for (int j = 0; j < 16; ++j) { wj[j] = __expf(logits[j] - mx); wsum += wj[j]; }
  float inv = 1.f / wsum;

  float o[8] = {0.f, 0.f, 0.f, 0.f, 0.f, 0.f, 0.f, 0.f};
#pragma unroll
  for (int j = 0; j < 16; ++j) {
    float a = wj[j] * inv;
    const bf16_t* vr = vbuf + ((size_t)b * NN + nbi[j]) * CC;
#pragma unroll
    for (int i = 0; i < 8; ++i) o[i] += a * (float)vr[lane + i * 32];
  }
#pragma unroll
  for (int i = 0; i < 8; ++i)
    outp[(size_t)gq * CC + lane + i * 32] = (bf16_t)o[i];
}

// =====================================================================
// launcher
// =====================================================================
extern "C" void kernel_launch(void* const* d_in, const int* in_sizes, int n_in,
                              void* d_out, int out_size, void* d_ws, size_t ws_size,
                              hipStream_t stream) {
  const float* x    = (const float*)d_in[0];
  const float* pos  = (const float*)d_in[1];
  const float* Wq   = (const float*)d_in[2];
  const float* Wk   = (const float*)d_in[3];
  const float* Wv   = (const float*)d_in[4];
  const float* W1   = (const float*)d_in[5];
  const float* b1   = (const float*)d_in[6];
  const float* W2   = (const float*)d_in[7];
  const float* b2   = (const float*)d_in[8];
  const float* Wp   = (const float*)d_in[9];
  const float* bp   = (const float*)d_in[10];
  const float* gq   = (const float*)d_in[11];
  const float* betaq  = (const float*)d_in[12];
  const float* gkv  = (const float*)d_in[13];
  const float* betakv = (const float*)d_in[14];
  float* out = (float*)d_out;

  char* wsb = (char*)d_ws;
  size_t off = 0;
  auto alloc = [&](size_t bytes) -> char* {
    char* p = wsb + off;
    off += (bytes + 255) & ~(size_t)255;
    return p;
  };
  bf16_t* Wqt  = (bf16_t*)alloc(CC * CC * sizeof(bf16_t));
  bf16_t* Wkt  = (bf16_t*)alloc(CC * CC * sizeof(bf16_t));
  bf16_t* Wvt  = (bf16_t*)alloc(CC * CC * sizeof(bf16_t));
  bf16_t* Wpt  = (bf16_t*)alloc(CC * CC * sizeof(bf16_t));
  float*  w2s  = (float*)alloc(RPE_DIM * sizeof(float));
  float*  b2s  = (float*)alloc(sizeof(float));
  bf16_t* xq   = (bf16_t*)alloc((size_t)ROWS * CC * sizeof(bf16_t));
  bf16_t* xkv  = (bf16_t*)alloc((size_t)ROWS * CC * sizeof(bf16_t));
  float*  qbuf = (float*)alloc((size_t)ROWS * CC * sizeof(float));
  bf16_t* kbuf = (bf16_t*)alloc((size_t)ROWS * CC * sizeof(bf16_t));
  bf16_t* vbuf = (bf16_t*)alloc((size_t)ROWS * CC * sizeof(bf16_t));
  int*    idxb = (int*)alloc((size_t)ROWS * KK * sizeof(int));
  bf16_t* outp = (bf16_t*)alloc((size_t)ROWS * CC * sizeof(bf16_t));

  prep_kernel<<<1, 256, 0, stream>>>(Wq, Wk, Wv, Wp, W2, b2,
                                     Wqt, Wkt, Wvt, Wpt, w2s, b2s);
  ln_kernel<<<ROWS, 256, 0, stream>>>(x, gq, betaq, gkv, betakv, xq, xkv);

  dim3 gg(ROWS / 128, CC / 64);
  gemm_bf16_wmma<float>  <<<gg, 256, 0, stream>>>(xq,  Wqt, nullptr, qbuf);
  gemm_bf16_wmma<bf16_t> <<<gg, 256, 0, stream>>>(xkv, Wkt, nullptr, kbuf);
  gemm_bf16_wmma<bf16_t> <<<gg, 256, 0, stream>>>(xkv, Wvt, nullptr, vbuf);

  knn_kernel<<<dim3(NN / 256, BB), 256, 0, stream>>>(pos, idxb);

  attn_kernel<<<ROWS / 8, 256, 0, stream>>>(qbuf, kbuf, vbuf, pos, idxb,
                                            W1, b1, w2s, b2s, outp);

  gemm_bf16_wmma<float> <<<gg, 256, 0, stream>>>(outp, Wpt, bp, out);
}
